// NNEmb_68427418960537
// MI455X (gfx1250) — matrix operand order repaired and checked by reading, not compiled, not run on hardware
//
#include <hip/hip_runtime.h>
#include <hip/hip_bf16.h>

typedef __attribute__((ext_vector_type(16))) __bf16 v16bf;
typedef __attribute__((ext_vector_type(8)))  float  v8f;
typedef __attribute__((ext_vector_type(8)))  unsigned short ushort8;

#define BQ  1024
#define DIM 512
#define EPS 1e-8f

__device__ __forceinline__ unsigned short f32_to_bf16_rne(float f) {
  unsigned int u = __float_as_uint(f);
  unsigned int r = u + 0x7FFFu + ((u >> 16) & 1u);   // round-to-nearest-even
  return (unsigned short)(r >> 16);
}

// Order-preserving float -> u32 key; pack with ~n so ties prefer smaller n.
__device__ __forceinline__ unsigned long long pack_key(float s, int n) {
  unsigned int b = __float_as_uint(s);
  unsigned int k = (b & 0x80000000u) ? ~b : (b | 0x80000000u);
  return ((unsigned long long)k << 32) | (unsigned int)(~(unsigned int)n);
}

__device__ __forceinline__ unsigned long long shfl_xor_u64(unsigned long long v, int mask) {
  unsigned int lo = (unsigned int)v;
  unsigned int hi = (unsigned int)(v >> 32);
  lo = __shfl_xor(lo, mask, 32);
  hi = __shfl_xor(hi, mask, 32);
  return ((unsigned long long)hi << 32) | lo;
}

// ---------------------------------------------------------------------------
// Kernel 0: zero the per-query best (packed score|idx) buffer.
// ---------------------------------------------------------------------------
__global__ void init_best_kernel(unsigned long long* __restrict__ best) {
  int q = blockIdx.x * blockDim.x + threadIdx.x;
  if (q < BQ) best[q] = 0ull;
}

// ---------------------------------------------------------------------------
// Kernel 1: query embeddings = mean over S token rows; emit bf16 Q + q-norm.
// One block per query; 256 threads, float2 per thread (512 floats / row).
// ---------------------------------------------------------------------------
__global__ __launch_bounds__(256)
void embed_mean_kernel(const int* __restrict__ tokens,
                       const float* __restrict__ emb,
                       unsigned short* __restrict__ Qb,
                       float* __restrict__ qn, int S) {
  __shared__ float red[256];
  const int q = blockIdx.x;
  const int t = threadIdx.x;
  float2 acc = make_float2(0.f, 0.f);
  for (int s = 0; s < S; ++s) {
    int tok = tokens[s * BQ + q];                       // uniform per block
    const float2* row = (const float2*)(emb + ((size_t)tok << 9));
    float2 v = row[t];
    acc.x += v.x; acc.y += v.y;
  }
  float inv = 1.0f / (float)S;
  acc.x *= inv; acc.y *= inv;
  unsigned int packed = (unsigned int)f32_to_bf16_rne(acc.x) |
                        ((unsigned int)f32_to_bf16_rne(acc.y) << 16);
  ((unsigned int*)(Qb + ((size_t)q << 9)))[t] = packed;
  red[t] = acc.x * acc.x + acc.y * acc.y;
  __syncthreads();
  for (int off = 128; off > 0; off >>= 1) {
    if (t < off) red[t] += red[t + off];
    __syncthreads();
  }
  if (t == 0) qn[q] = sqrtf(red[0]);
}

// ---------------------------------------------------------------------------
// Kernel 2: train matrix fp32 -> bf16 (single HBM stream) + t-norms.
// One block per train row.
// ---------------------------------------------------------------------------
__global__ __launch_bounds__(256)
void train_prep_kernel(const float* __restrict__ T,
                       unsigned short* __restrict__ Tb,
                       float* __restrict__ tn) {
  __shared__ float red[256];
  const int n = blockIdx.x;
  const int t = threadIdx.x;
  const float2* row = (const float2*)(T + ((size_t)n << 9));
  float2 v = row[t];
  unsigned int packed = (unsigned int)f32_to_bf16_rne(v.x) |
                        ((unsigned int)f32_to_bf16_rne(v.y) << 16);
  ((unsigned int*)(Tb + ((size_t)n << 9)))[t] = packed;
  red[t] = v.x * v.x + v.y * v.y;
  __syncthreads();
  for (int off = 128; off > 0; off >>= 1) {
    if (t < off) red[t] += red[t + off];
    __syncthreads();
  }
  if (t == 0) tn[n] = sqrtf(red[0]);
}

// ---------------------------------------------------------------------------
// Kernel 3: bf16 WMMA GEMM (Q . T^T) fused with cosine scaling + argmax merge.
// Wave tile: 32(M) x 64(N): 2 A-frags x 4 B-frags -> 8 WMMAs per 12 b128
// loads (1.5 loads/WMMA).  Block: 8 waves as 2Mx4N -> 64M x 256N.
// Grid.x = M tiles (fastest) so consecutive blocks reuse the same B panel.
// ---------------------------------------------------------------------------
__global__ __launch_bounds__(256)
void gemm_argmax_kernel(const unsigned short* __restrict__ Qb,
                        const unsigned short* __restrict__ Tb,
                        const float* __restrict__ qn,
                        const float* __restrict__ tn,
                        unsigned long long* __restrict__ best,
                        int N) {
  const int tid  = threadIdx.x;
  const int lane = tid & 31;
  const int w    = tid >> 5;
  const int wm   = w & 1;          // 2 M sub-tiles of 32
  const int wn   = w >> 1;         // 4 N sub-tiles of 64
  const int m0   = blockIdx.x * 64 + wm * 32;
  const int n0   = blockIdx.y * 256 + wn * 64;

  const int r    = lane & 15;      // row-in-tile / column-in-tile
  const int half = lane >> 4;      // 0: lanes 0-15, 1: lanes 16-31
  const int hk   = half << 3;      // A-fragment K sub-offset (0 / 8)
  const int kk   = half << 4;      // unused (0 / 16); B loads use hk

  // A fragment sources: query rows m0+r and m0+16+r, K-contiguous bf16.
  const unsigned short* qrow0 = Qb + (((size_t)(m0 + r)) << 9);
  const unsigned short* qrow1 = Qb + (((size_t)(m0 + 16 + r)) << 9);

  // B fragment sources: 4 train rows per lane (columns of B), clamped at N.
  const unsigned short* trow[4];
  int   nidx[4];
  bool  nvalid[4];
  float tnv[4];
#pragma unroll
  for (int f = 0; f < 4; ++f) {
    int n = n0 + f * 16 + r;
    nvalid[f] = (n < N);
    int nc = nvalid[f] ? n : 0;
    nidx[f]  = n;
    trow[f]  = Tb + (((size_t)nc) << 9);
    tnv[f]   = tn[nc];
  }

  v8f c[2][4] = {};                 // fp32 accumulators (64 VGPRs)

#pragma unroll 4
  for (int k0 = 0; k0 < DIM; k0 += 32) {
    union { ushort8 u[2]; v16bf v; } A0, A1;
    A0.u[0] = *(const ushort8*)(qrow0 + k0 + hk);        // K {0-7 | 8-15}
    A0.u[1] = *(const ushort8*)(qrow0 + k0 + 16 + hk);   // K {16-23 | 24-31}
    A1.u[0] = *(const ushort8*)(qrow1 + k0 + hk);
    A1.u[1] = *(const ushort8*)(qrow1 + k0 + 16 + hk);
#pragma unroll
    for (int f = 0; f < 4; ++f) {
      union { ushort8 u[2]; v16bf v; } B;
      B.u[0] = *(const ushort8*)(trow[f] + k0 + hk);       // K {0-7 | 8-15}
      B.u[1] = *(const ushort8*)(trow[f] + k0 + 16 + hk);  // K {16-23 | 24-31}
      c[0][f] = __builtin_amdgcn_wmma_f32_16x16x32_bf16(
          false, A0.v, false, B.v, (short)0, c[0][f], false, false);
      c[1][f] = __builtin_amdgcn_wmma_f32_16x16x32_bf16(
          false, A1.v, false, B.v, (short)0, c[1][f], false, false);
    }
  }

  // Epilogue: scale by 1/max(|q||t|, eps), reduce argmax in-wave, merge.
  // C layout: VGPR i of c[g][f] holds (M = m0 + g*16 + i + 8*half, N = col).
#pragma unroll
  for (int g = 0; g < 2; ++g) {
#pragma unroll
    for (int i = 0; i < 8; ++i) {
      int m = m0 + g * 16 + i + (half << 3);
      float qnm = qn[m];
      unsigned long long kv = 0ull;
#pragma unroll
      for (int f = 0; f < 4; ++f) {
        float denom = fmaxf(qnm * tnv[f], EPS);
        float sim = c[g][f][i] / denom;
        unsigned long long key = nvalid[f] ? pack_key(sim, nidx[f]) : 0ull;
        kv = (key > kv) ? key : kv;
      }
      // max over the 16 lanes of each wave half (xor<16 stays in-half)
#pragma unroll
      for (int mask = 1; mask <= 8; mask <<= 1) {
        unsigned long long o = shfl_xor_u64(kv, mask);
        kv = (o > kv) ? o : kv;
      }
      if (r == 0) atomicMax(best + m, kv);
    }
  }
}

// ---------------------------------------------------------------------------
// Kernel 4: unpack best -> (score, y_train[idx]) into d_out.
// ---------------------------------------------------------------------------
__global__ void finalize_kernel(const unsigned long long* __restrict__ best,
                                const int* __restrict__ y_train,
                                float* __restrict__ out, int N) {
  int q = blockIdx.x * blockDim.x + threadIdx.x;
  if (q >= BQ) return;
  unsigned long long kv = best[q];
  unsigned int k = (unsigned int)(kv >> 32);
  unsigned int b = (k & 0x80000000u) ? (k & 0x7FFFFFFFu) : ~k;
  unsigned int idx = ~(unsigned int)(kv & 0xFFFFFFFFu);
  if (idx >= (unsigned int)N) idx = 0;
  out[q]      = __uint_as_float(b);
  out[BQ + q] = (float)y_train[idx];
}

// ---------------------------------------------------------------------------
extern "C" void kernel_launch(void* const* d_in, const int* in_sizes, int n_in,
                              void* d_out, int out_size, void* d_ws, size_t ws_size,
                              hipStream_t stream) {
  const int*   tokens  = (const int*)d_in[0];     // [S, 1024] int32
  const float* emb     = (const float*)d_in[1];   // [V, 512]  fp32
  const float* train   = (const float*)d_in[2];   // [N, 512]  fp32
  const int*   y_train = (const int*)d_in[3];     // [N]       int32
  const int S = in_sizes[0] / BQ;                 // 128
  const int N = in_sizes[3];                      // 100000
  float* out = (float*)d_out;

  // Workspace layout (~104 MB)
  char* ws = (char*)d_ws;
  size_t off = 0;
  unsigned short* Qb = (unsigned short*)(ws + off); off += (size_t)BQ * DIM * 2;
  unsigned short* Tb = (unsigned short*)(ws + off); off += (size_t)N  * DIM * 2;
  off = (off + 255) & ~(size_t)255;
  float* qn = (float*)(ws + off); off += (size_t)BQ * 4;
  float* tn = (float*)(ws + off); off += (size_t)N * 4;
  off = (off + 255) & ~(size_t)255;
  unsigned long long* best = (unsigned long long*)(ws + off); off += (size_t)BQ * 8;

  init_best_kernel<<<(BQ + 255) / 256, 256, 0, stream>>>(best);
  embed_mean_kernel<<<BQ, 256, 0, stream>>>(tokens, emb, Qb, qn, S);
  train_prep_kernel<<<N, 256, 0, stream>>>(train, Tb, tn);
  dim3 grid(BQ / 64, (N + 255) / 256);            // M fastest -> B-panel L2 reuse
  gemm_argmax_kernel<<<grid, 256, 0, stream>>>(Qb, Tb, qn, tn, best, N);
  finalize_kernel<<<(BQ + 255) / 256, 256, 0, stream>>>(best, y_train, out, N);
}
